// MPNNConvModel_15264313770095
// MI455X (gfx1250) — matrix operand order repaired and checked by reading, not compiled, not run on hardware
//
#include <hip/hip_runtime.h>
#include <math.h>

#define N_USERS 100000
#define N_ITEMS 50000
#define N_EDGES 1000000
#define N_PAIRS 500000
#define D_OUT_  64
#define EPS 1e-12f
#define LDA 280   // bf16 elems per LDS A row: 16B-aligned runs, conflict-free banks

typedef __attribute__((ext_vector_type(16))) __bf16 v16bf;
typedef __attribute__((ext_vector_type(8)))  float  v8f;

union BFrag { v16bf v; unsigned short h[16]; unsigned u[8]; uint4 q[2]; };

__device__ __forceinline__ unsigned short f2bf(float f) {
  union { float f; unsigned u; } x; x.f = f;
  unsigned u = x.u;
  u += 0x7FFFu + ((u >> 16) & 1u);           // round-to-nearest-even
  return (unsigned short)(u >> 16);
}

// pack two f32 -> packed bf16 pair (x -> low 16, y -> high 16)
__device__ __forceinline__ unsigned pk_bf16(float x, float y) {
#if __has_builtin(__builtin_amdgcn_cvt_pk_bf16_f32)
  typedef __attribute__((ext_vector_type(2))) __bf16 v2bf;
  union { v2bf v; unsigned u; } c;
  c.v = __builtin_amdgcn_cvt_pk_bf16_f32(x, y);
  return c.u;
#else
  return (unsigned)f2bf(x) | ((unsigned)f2bf(y) << 16);
#endif
}

__device__ __forceinline__ unsigned lds_addr_of(const void* p) {
  return (unsigned)(unsigned long long)
      (__attribute__((address_space(3))) const void*)p;
}

__device__ __forceinline__ void async_cp16(const void* g, const void* lds) {
  // GLOBAL_LOAD_ASYNC_TO_LDS_B128: per-lane 16B global -> LDS, tracked by ASYNCcnt
  asm volatile("global_load_async_to_lds_b128 %0, %1, off"
               :: "v"(lds_addr_of(lds)), "v"((unsigned long long)g) : "memory");
}

__device__ __forceinline__ void wait_async0() {
#if __has_builtin(__builtin_amdgcn_s_wait_asynccnt)
  __builtin_amdgcn_s_wait_asynccnt(0);
#else
  asm volatile("s_wait_asynccnt 0x0" ::: "memory");
#endif
}

// ---------------- edge aggregation: one wave per edge -------------------
__global__ __launch_bounds__(256) void agg_kernel(
    const float* __restrict__ hsrc, const int* __restrict__ src,
    const int* __restrict__ dst, int nE,
    float* __restrict__ acc, float* __restrict__ deg) {
  int w    = (blockIdx.x * blockDim.x + threadIdx.x) >> 5;
  int lane = threadIdx.x & 31;
  if (w >= nE) return;
  int s = src[w], d = dst[w];
  float4 v = ((const float4*)(hsrc + (long long)s * 128))[lane];
  float* a = acc + (long long)d * 128 + lane * 4;
  atomicAdd(a + 0, v.x);
  atomicAdd(a + 1, v.y);
  atomicAdd(a + 2, v.z);
  atomicAdd(a + 3, v.w);
  if (lane == 0) atomicAdd(deg + d, 1.0f);
}

// ------ finalize: apk[row][0:128]=acc/deg, [128:256]=self, as bf16 ------
__global__ __launch_bounds__(256) void pack_kernel(
    const float* __restrict__ acc, const float* __restrict__ deg,
    const float* __restrict__ hdst, int n, unsigned short* __restrict__ apk) {
  int t = blockIdx.x * blockDim.x + threadIdx.x;   // 4 elements per thread
  if (t >= n * 64) return;
  int row = t >> 6;
  int q   = (t & 63) * 4;
  float4 v;
  if (q < 128) {
    float inv = 1.0f / fmaxf(deg[row], 1.0f);
    float4 a = *(const float4*)(acc + (long long)row * 128 + q);
    v = make_float4(a.x * inv, a.y * inv, a.z * inv, a.w * inv);
  } else {
    v = *(const float4*)(hdst + (long long)row * 128 + (q - 128));
  }
  *(uint2*)(apk + (long long)row * 256 + q) =
      make_uint2(pk_bf16(v.x, v.y), pk_bf16(v.z, v.w));
}

// ---- fused WMMA conv: A=[mean|self](bf16) @ [Wn;Ws], +b, ReLU, L2 ------
__global__ __launch_bounds__(256) void conv_kernel(
    const unsigned short* __restrict__ apk,
    const float* __restrict__ Wn, const float* __restrict__ bn,
    const float* __restrict__ Ws, const float* __restrict__ bs,
    int nDst, int nOut, float* __restrict__ out) {
  __shared__ unsigned short Ab[2][16 * LDA];
  __shared__ float rowsq[16];

  const int tid  = threadIdx.x;
  const int wave = tid >> 5;
  const int lane = tid & 31;
  const int half = lane >> 4;
  const int ln   = lane & 15;
  const int col  = wave * 16 + ln;

  // B preload: combined [Wn(128);Ws(128)] x nOut as bf16 fragments, block-resident.
  // B k order used here (contiguous, same as A): lanes 0-15 K=0..15, lanes 16-31 K=16..31, N=lane&15.
  BFrag bfr[8];
  #pragma unroll
  for (int kc = 0; kc < 8; ++kc) {
    const float* Wsrc = (kc < 4) ? Wn : Ws;
    int kb = (kc & 3) * 32 + half * 16;
    #pragma unroll
    for (int i = 0; i < 16; i += 2)
      bfr[kc].u[i >> 1] = pk_bf16(Wsrc[(kb + i) * nOut + col],
                                  Wsrc[(kb + i + 1) * nOut + col]);
  }
  const float bc = bn[col] + bs[col];

  const int nTiles = nDst >> 4;     // nDst is a multiple of 16 (50000/100000)

  // async prefetch of one 16x256-bf16 tile (512 x 16B chunks) into LDS buf
  auto issue_tile = [&](int tile, int b) {
    const unsigned short* gbase = apk + (long long)tile * 16 * 256;
    for (int c = tid; c < 512; c += blockDim.x) {
      int row = c >> 5, cb = c & 31;                    // 32 chunks per row
      async_cp16(gbase + row * 256 + cb * 8, &Ab[b][row * LDA + cb * 8]);
    }
  };

  int tile = blockIdx.x, buf = 0;
  if (tile < nTiles) issue_tile(tile, 0);

  for (; tile < nTiles; tile += gridDim.x) {
    wait_async0();
    __syncthreads();                       // b1: cur tile resident; prev reads done
    if (tid < 16) rowsq[tid] = 0.0f;
    int nxt = tile + gridDim.x;
    if (nxt < nTiles) issue_tile(nxt, buf ^ 1);   // overlap copy with MMA
    __syncthreads();                       // b2: rowsq zeros visible

    // Load all 8 A fragments first (16 pipelined ds_load_b128), then run the
    // 8 v_wmma_f32_16x16x32_bf16 back-to-back as one XDL accumulate chain.
    BFrag a[8];
    const unsigned short* abase = &Ab[buf][ln * LDA + half * 16];
    #pragma unroll
    for (int kc = 0; kc < 8; ++kc) {
      const unsigned short* ap = abase + kc * 32;
      a[kc].q[0] = *(const uint4*)(ap);
      a[kc].q[1] = *(const uint4*)(ap + 8);
    }
    v8f c = {};
    #pragma unroll
    for (int kc = 0; kc < 8; ++kc)
      c = __builtin_amdgcn_wmma_f32_16x16x32_bf16(false, a[kc].v, false, bfr[kc].v,
                                                  (short)0, c, false, false);

    // bias + ReLU; row sums of squares via value-halving butterfly (8 shuffles)
    float sq[8];
    #pragma unroll
    for (int r = 0; r < 8; ++r) {
      float z = fmaxf(c[r] + bc, 0.0f);
      c[r] = z;
      sq[r] = z * z;
    }
    {
      const bool h1 = ln & 1, h2 = ln & 2, h4 = ln & 4;
      #pragma unroll
      for (int i = 0; i < 4; ++i) {               // 16 lanes x 8 vals -> 8 x 4
        float give = h1 ? sq[i] : sq[i + 4];
        float keep = h1 ? sq[i + 4] : sq[i];
        sq[i] = keep + __shfl_xor(give, 1, 32);
      }
      #pragma unroll
      for (int i = 0; i < 2; ++i) {               // -> 4 x 2
        float give = h2 ? sq[i] : sq[i + 2];
        float keep = h2 ? sq[i + 2] : sq[i];
        sq[i] = keep + __shfl_xor(give, 2, 32);
      }
      {
        float give = h4 ? sq[0] : sq[1];          // -> 2 x 1
        float keep = h4 ? sq[1] : sq[0];
        sq[0] = keep + __shfl_xor(give, 4, 32);
      }
      sq[0] += __shfl_xor(sq[0], 8, 32);          // -> 1 x 1 (dup in ln^8)
      int rsel = ((ln & 1) << 2) | (ln & 2) | ((ln & 4) >> 2);
      if (ln < 8)
        atomicAdd(&rowsq[rsel + half * 8], sq[0]);  // distinct addrs, no conflicts
    }
    __syncthreads();                       // b3: rowsq complete across waves

    const int rowBase = tile << 4;
    #pragma unroll
    for (int r = 0; r < 8; ++r) {
      float scale = 1.0f / fmaxf(sqrtf(rowsq[r + half * 8]), EPS);
      out[(long long)(rowBase + r + half * 8) * nOut + col] = c[r] * scale;
    }
    buf ^= 1;
  }
}

// ---------------- cosine scores: one wave per pair ----------------------
__global__ __launch_bounds__(256) void cos_kernel(
    const float* __restrict__ hu, const float* __restrict__ hi,
    const int* __restrict__ pu, const int* __restrict__ pi,
    const int* __restrict__ qu, const int* __restrict__ qi,
    float* __restrict__ posOut, float* __restrict__ negOut) {
  int w    = (blockIdx.x * blockDim.x + threadIdx.x) >> 5;
  int lane = threadIdx.x & 31;
  if (w >= 2 * N_PAIRS) return;
  int u, it; float* o;
  if (w < N_PAIRS) { u = pu[w]; it = pi[w]; o = posOut + w; }
  else             { int p = w - N_PAIRS; u = qu[p]; it = qi[p]; o = negOut + p; }
  float2 a = ((const float2*)(hu + (long long)u  * 64))[lane];
  float2 b = ((const float2*)(hi + (long long)it * 64))[lane];
  float dot = a.x * b.x + a.y * b.y;
  float su  = a.x * a.x + a.y * a.y;
  float si  = b.x * b.x + b.y * b.y;
  for (int off = 1; off < 32; off <<= 1) {
    dot += __shfl_xor(dot, off, 32);
    su  += __shfl_xor(su,  off, 32);
    si  += __shfl_xor(si,  off, 32);
  }
  if (lane == 0)
    *o = dot / (fmaxf(sqrtf(su), EPS) * fmaxf(sqrtf(si), EPS));
}

extern "C" void kernel_launch(void* const* d_in, const int* in_sizes, int n_in,
                              void* d_out, int out_size, void* d_ws, size_t ws_size,
                              hipStream_t stream) {
  (void)in_sizes; (void)n_in; (void)out_size; (void)ws_size;
  const float* h_user  = (const float*)d_in[0];
  const float* h_item  = (const float*)d_in[1];
  const float* Wn01    = (const float*)d_in[2];
  const float* bn01    = (const float*)d_in[3];
  const float* Ws01    = (const float*)d_in[4];
  const float* bs01    = (const float*)d_in[5];
  const float* Wn2     = (const float*)d_in[6];
  const float* bn2     = (const float*)d_in[7];
  const float* Ws2     = (const float*)d_in[8];
  const float* bs2     = (const float*)d_in[9];
  const int* u2i_src = (const int*)d_in[10];
  const int* u2i_dst = (const int*)d_in[11];
  const int* i2u_src = (const int*)d_in[12];
  const int* i2u_dst = (const int*)d_in[13];
  const int* pos_u   = (const int*)d_in[14];
  const int* pos_i   = (const int*)d_in[15];
  const int* neg_u   = (const int*)d_in[16];
  const int* neg_i   = (const int*)d_in[17];

  // workspace layout
  char* w = (char*)d_ws;
  float* acc = (float*)w;            w += (size_t)N_USERS * 128 * 4;
  float* deg = (float*)w;            w += (size_t)N_USERS * 4;
  unsigned short* apk = (unsigned short*)w;  w += (size_t)N_USERS * 256 * 2;
  float* huA = (float*)w;            w += (size_t)N_USERS * 128 * 4;
  float* huB = (float*)w;            w += (size_t)N_USERS * 128 * 4;
  float* hiA = (float*)w;            w += (size_t)N_ITEMS * 128 * 4;
  float* hiB = (float*)w;

  float* out     = (float*)d_out;
  float* hu_out  = out;                                   // N_USERS*64
  float* hi_out  = out + (size_t)N_USERS * D_OUT_;        // N_ITEMS*64
  float* pos_out = hi_out + (size_t)N_ITEMS * D_OUT_;     // N_PAIRS
  float* neg_out = pos_out + N_PAIRS;                     // N_PAIRS

  const int aggBlocks = (N_EDGES + 7) / 8;   // 8 edge-waves per 256-thread block

  auto run_conv = [&](const float* hsrc, const float* hdst_,
                      const int* esrc, const int* edst, int nDst,
                      const float* Wn, const float* bn_, const float* Ws_,
                      const float* bs_, int nOut, float* outp) {
    hipMemsetAsync(acc, 0, (size_t)nDst * 128 * sizeof(float), stream);
    hipMemsetAsync(deg, 0, (size_t)nDst * sizeof(float), stream);
    agg_kernel<<<aggBlocks, 256, 0, stream>>>(hsrc, esrc, edst, N_EDGES, acc, deg);
    pack_kernel<<<(nDst * 64 + 255) / 256, 256, 0, stream>>>(acc, deg, hdst_, nDst, apk);
    int nTiles = nDst / 16;
    int blocks = nTiles < 1024 ? nTiles : 1024;   // >1 tile/block -> async overlap
    conv_kernel<<<blocks, nOut * 2, 0, stream>>>(apk, Wn, bn_, Ws_, bs_,
                                                 nDst, nOut, outp);
  };

  const float* hu = h_user;
  const float* hi = h_item;
  // layer 0  (weights [l*2+e]; e=0: user->item, e=1: item->user)
  run_conv(hu, hi, u2i_src, u2i_dst, N_ITEMS,
           Wn01 + 0 * 128 * 128, bn01 + 0 * 128, Ws01 + 0 * 128 * 128, bs01 + 0 * 128, 128, hiA);
  run_conv(hi, hu, i2u_src, i2u_dst, N_USERS,
           Wn01 + 1 * 128 * 128, bn01 + 1 * 128, Ws01 + 1 * 128 * 128, bs01 + 1 * 128, 128, huA);
  hu = huA; hi = hiA;
  // layer 1
  run_conv(hu, hi, u2i_src, u2i_dst, N_ITEMS,
           Wn01 + 2 * 128 * 128, bn01 + 2 * 128, Ws01 + 2 * 128 * 128, bs01 + 2 * 128, 128, hiB);
  run_conv(hi, hu, i2u_src, i2u_dst, N_USERS,
           Wn01 + 3 * 128 * 128, bn01 + 3 * 128, Ws01 + 3 * 128 * 128, bs01 + 3 * 128, 128, huB);
  hu = huB; hi = hiB;
  // layer 2 (nOut=64) -> straight into d_out
  run_conv(hu, hi, u2i_src, u2i_dst, N_ITEMS,
           Wn2 + 0 * 128 * 64, bn2 + 0 * 64, Ws2 + 0 * 128 * 64, bs2 + 0 * 64, 64, hi_out);
  run_conv(hi, hu, i2u_src, i2u_dst, N_USERS,
           Wn2 + 1 * 128 * 64, bn2 + 1 * 64, Ws2 + 1 * 128 * 64, bs2 + 1 * 64, 64, hu_out);

  cos_kernel<<<(2 * N_PAIRS + 7) / 8, 256, 0, stream>>>(hu_out, hi_out,
                                                        pos_u, pos_i, neg_u, neg_i,
                                                        pos_out, neg_out);
}
